// TripleAdaptiveQuantizerV13_Eval_89781996356256
// MI455X (gfx1250) — matrix operand order repaired and checked by reading, not compiled, not run on hardware
//
#include <hip/hip_runtime.h>
#include <math.h>

// ---------------- problem constants (fixed by the reference) ----------------
#define B_  16
#define L_  1024
#define K_  8
#define N_  64
#define HH  64
#define NTOK (B_*L_*K_)      // 131072 tokens
#define NBL  (B_*L_)         // 16384 (b,l) pairs

typedef unsigned short ushort_t;
typedef __attribute__((ext_vector_type(16))) __bf16 v16bf;
typedef __attribute__((ext_vector_type(8)))  float  v8f;
typedef __attribute__((ext_vector_type(4)))  unsigned int v4u;
typedef union { v4u u[2]; v16bf b; } bfu;

// ---------------- workspace layout (bytes) ----------------
#define WS_HP   ((size_t)0)          // h_power f32, NTOK        (524288)
#define WS_AVG  ((size_t)524288)     // avg sums f32, B_*K_      (512, pad 1024)
#define WS_YPW  ((size_t)525312)     // log1p(y power) f32, NBL  (65536)
#define WS_X    ((size_t)590848)     // X bf16 NTOK x 32         (8388608)
#define WS_W1S  ((size_t)8979456)    // swizzled W1 frags        (4096)
#define WS_W2S  ((size_t)8983552)    // swizzled W2 frags        (8192)
#define WS_W3S  ((size_t)8991744)    // swizzled W3 frags        (8192)
#define WS_WHS  ((size_t)8999936)    // swizzled head frags      (2048)
#define WS_IY   ((size_t)9001984)    // int32 NBL                (65536)
#define WS_IH   ((size_t)9067520)    // int32 NTOK               (524288)
#define WS_ID   ((size_t)9591808)    // int32 NTOK               (524288)
// total ~10.1 MB

// ---------------- output layout (f32 elements) ----------------
#define OFF_VQ  ((size_t)0)
#define OFF_HQ  ((size_t)262144)
#define OFF_YQ  ((size_t)17039360)
#define OFF_TOT ((size_t)19136512)
#define OFF_BY  ((size_t)19267584)
#define OFF_BH  ((size_t)19398656)
#define OFF_BD  ((size_t)19529728)

__device__ __forceinline__ ushort_t f2bf(float f) {
  unsigned u = __float_as_uint(f);
  unsigned r = u + 0x7FFFu + ((u >> 16) & 1u);   // RNE truncate to bf16
  return (ushort_t)(r >> 16);
}

// round(clip(x/s, qn, qp)) * s  ; idx==0 -> 0 ; bits = 4*idx
__device__ __forceinline__ float quant1(float x, int idx, const float* svec) {
  if (idx == 0) return 0.f;
  float s  = svec[idx - 1];
  float qp = (float)((1 << (4 * idx - 1)) - 1);
  float qn = -qp - 1.f;
  float xs = fminf(fmaxf(x / s, qn), qp);
  return rintf(xs) * s;
}

// ============================================================================
// K0: zero avg accumulators + pre-swizzle all weights into WMMA B-fragment
// order: Wsw[((frag*32 + lane)*16 + j)] = W[k][n] with
//   n = ntile*16 + lane%16 ; k = kstep*32 + j + 16*(lane/16)
// so at run time each lane does ONE contiguous 32-byte load per fragment.
// ============================================================================
__global__ void k0_prep(const float* W1, const float* W2, const float* W3,
                        const float* Wy, const float* Wh, const float* Wd,
                        float* avg, ushort_t* W1s, ushort_t* W2s,
                        ushort_t* W3s, ushort_t* WHs) {
  int t = threadIdx.x;
  if (t < B_ * K_) avg[t] = 0.f;
  // W1 padded to 32x64 : 4 frags (ntile 0..3, kstep 0)
  for (int idx = t; idx < 4 * 512; idx += 256) {
    int f = idx >> 9, lam = (idx >> 4) & 31, j = idx & 15;
    int n = f * 16 + (lam & 15);
    int k = j + 16 * (lam >> 4);
    float v = (k < 9) ? W1[k * 64 + n] : 0.f;
    W1s[idx] = f2bf(v);
  }
  // W2 64x64 : 8 frags (f = ntile*2 + kstep)
  for (int idx = t; idx < 8 * 512; idx += 256) {
    int f = idx >> 9, lam = (idx >> 4) & 31, j = idx & 15;
    int nt = f >> 1, ks = f & 1;
    int n = nt * 16 + (lam & 15);
    int k = ks * 32 + j + 16 * (lam >> 4);
    W2s[idx] = f2bf(W2[k * 64 + n]);
  }
  for (int idx = t; idx < 8 * 512; idx += 256) {
    int f = idx >> 9, lam = (idx >> 4) & 31, j = idx & 15;
    int nt = f >> 1, ks = f & 1;
    int n = nt * 16 + (lam & 15);
    int k = ks * 32 + j + 16 * (lam >> 4);
    W3s[idx] = f2bf(W3[k * 64 + n]);
  }
  // heads packed 64x16 : cols 0-4 Wy, 5-9 Wh, 10-14 Wd, 15 zero. 2 frags (kstep)
  for (int idx = t; idx < 2 * 512; idx += 256) {
    int f = idx >> 9, lam = (idx >> 4) & 31, j = idx & 15;
    int n = lam & 15;
    int k = f * 32 + j + 16 * (lam >> 4);
    float v = 0.f;
    if      (n < 5)  v = Wy[k * 5 + n];
    else if (n < 10) v = Wh[k * 5 + (n - 5)];
    else if (n < 15) v = Wd[k * 5 + (n - 10)];
    WHs[idx] = f2bf(v);
  }
}

// ============================================================================
// K1: per (b,l): h_power[k] = sum_{n,c} H^2 ; y_power = log1p(sum y^2)
//     + atomic accumulation of per-(b,k) sums for avg_cp.
// ============================================================================
__global__ void k1_power(const float* __restrict__ H, const float* __restrict__ y,
                         float* __restrict__ hp, float* __restrict__ avg,
                         float* __restrict__ ypw) {
  int bl = blockIdx.x;
  const float* Hb = H + (size_t)bl * 1024;
  __shared__ float part[128];
  __shared__ float ys[128];
  int t = threadIdx.x, k = t & 7, g = t >> 3;
  float s = 0.f;
  for (int i = 0; i < 8; ++i) {
    int e = g * 8 + i;            // e in [0,128): n = e>>1, c = e&1
    int n = e >> 1, c = e & 1;
    float x = Hb[(n * 8 + k) * 2 + c];
    s += x * x;
  }
  part[k * 16 + g] = s;
  float yv = y[(size_t)bl * 128 + t];
  ys[t] = yv * yv;
  __syncthreads();
  if (t < 8) {
    float acc = 0.f;
    for (int g2 = 0; g2 < 16; ++g2) acc += part[t * 16 + g2];
    hp[(size_t)bl * 8 + t] = acc;
    atomicAdd(&avg[(bl >> 10) * 8 + t], acc);
  }
  if (t == 0) {
    float a = 0.f;
    for (int i = 0; i < 128; ++i) a += ys[i];
    ypw[bl] = log1pf(a);
  }
}

// ============================================================================
// K2: build bf16 feature rows X[tok][0..31] (9 real features, rest zero)
// ============================================================================
__global__ void k2_features(const float* __restrict__ v, const float* __restrict__ snr,
                            const float* __restrict__ hp, const float* __restrict__ avg,
                            const float* __restrict__ ypw, ushort_t* __restrict__ X) {
  int tok = blockIdx.x * 256 + threadIdx.x;
  int k = tok & 7, bl = tok >> 3, b = bl >> 10;
  float h = hp[tok];
  const float* hr = hp + (size_t)bl * 8;
  float tp = 0.f;
  for (int i = 0; i < 8; ++i) tp += hr[i];
  float vr = v[(size_t)tok * 2], vi = v[(size_t)tok * 2 + 1];
  float feats[9] = {
    vr, vi, snr[tok], h, avg[b * 8 + k] * (1.f / (float)L_),
    log1pf(h / (tp - h + 1e-10f)), log1pf(tp), ypw[bl],
    sqrtf(vr * vr + vi * vi + 1e-10f)
  };
  __attribute__((aligned(16))) ushort_t row[32];
  for (int i = 0; i < 32; ++i) row[i] = (i < 9) ? f2bf(feats[i]) : (ushort_t)0;
  uint4* dst = (uint4*)(X + (size_t)tok * 32);
  const uint4* src = (const uint4*)row;
  dst[0] = src[0]; dst[1] = src[1]; dst[2] = src[2]; dst[3] = src[3];
}

// ============================================================================
// K3: fused MLP via v_wmma_f32_16x16x32_bf16. 8 waves/block, each wave owns a
// 16-token strip: X(16x32) -> h1(16x64) -> h2 -> h3 -> heads(16x15), then
// gumbel-hard argmax -> index buffers iy (per b,l), ih/id (per token).
// Activations ping-pong through LDS in row-major bf16; A/B fragments follow
// a contiguous k order (k = 16*(lane/16) + j) in both operands.
// ============================================================================
__global__ __launch_bounds__(256) void k3_mlp(
    const ushort_t* __restrict__ X,
    const ushort_t* __restrict__ W1s, const ushort_t* __restrict__ W2s,
    const ushort_t* __restrict__ W3s, const ushort_t* __restrict__ WHs,
    const float* __restrict__ b1, const float* __restrict__ b2,
    const float* __restrict__ b3, const float* __restrict__ by,
    const float* __restrict__ bh, const float* __restrict__ bd,
    const float* __restrict__ gy, const float* __restrict__ gH,
    const float* __restrict__ gd,
    int* __restrict__ iy, int* __restrict__ ih, int* __restrict__ id) {
  __shared__ ushort_t actA[8][16 * 64];
  __shared__ ushort_t actB[8][16 * 64];
  __shared__ float    lgt[8][16 * 16];

  int w   = threadIdx.x >> 5;
  int lam = threadIdx.x & 31;
  int s   = blockIdx.x * 8 + w;
  int lo  = lam & 15;          // A: row M ; B/D: col N
  int hi  = lam >> 4;

  auto loadB = [&](const ushort_t* Ws, int f) -> v16bf {
    return *(const v16bf*)(Ws + ((size_t)(f * 32 + lam)) * 16);
  };
  auto loadA = [&](const ushort_t* base, int stride, int kofs) -> v16bf {
    const ushort_t* p = base + lo * stride + kofs + hi * 16;
    bfu u;
    u.u[0] = *(const v4u*)p;         // K = kofs + 16*hi .. +7
    u.u[1] = *(const v4u*)(p + 8);   // K = kofs + 16*hi + 8 .. +7
    return u.b;
  };

  // ---- layer 1: X (K=32, padded) -> actA ----
  {
    v16bf a0 = loadA(X + (size_t)s * 16 * 32, 32, 0);
    for (int nt = 0; nt < 4; ++nt) {
      v8f acc = {};
      acc = __builtin_amdgcn_wmma_f32_16x16x32_bf16(false, a0, false,
              loadB(W1s, nt), (short)0, acc, false, false);
      float bias = b1[nt * 16 + lo];
      for (int r = 0; r < 8; ++r) {
        float vv = fmaxf(acc[r] + bias, 0.f);
        actA[w][(r + hi * 8) * 64 + nt * 16 + lo] = f2bf(vv);
      }
    }
  }
  // ---- layer 2: actA -> actB ----
  {
    v16bf a0 = loadA(actA[w], 64, 0);
    v16bf a1 = loadA(actA[w], 64, 32);
    for (int nt = 0; nt < 4; ++nt) {
      v8f acc = {};
      acc = __builtin_amdgcn_wmma_f32_16x16x32_bf16(false, a0, false,
              loadB(W2s, nt * 2 + 0), (short)0, acc, false, false);
      acc = __builtin_amdgcn_wmma_f32_16x16x32_bf16(false, a1, false,
              loadB(W2s, nt * 2 + 1), (short)0, acc, false, false);
      float bias = b2[nt * 16 + lo];
      for (int r = 0; r < 8; ++r) {
        float vv = fmaxf(acc[r] + bias, 0.f);
        actB[w][(r + hi * 8) * 64 + nt * 16 + lo] = f2bf(vv);
      }
    }
  }
  // ---- layer 3: actB -> actA ----
  {
    v16bf a0 = loadA(actB[w], 64, 0);
    v16bf a1 = loadA(actB[w], 64, 32);
    for (int nt = 0; nt < 4; ++nt) {
      v8f acc = {};
      acc = __builtin_amdgcn_wmma_f32_16x16x32_bf16(false, a0, false,
              loadB(W3s, nt * 2 + 0), (short)0, acc, false, false);
      acc = __builtin_amdgcn_wmma_f32_16x16x32_bf16(false, a1, false,
              loadB(W3s, nt * 2 + 1), (short)0, acc, false, false);
      float bias = b3[nt * 16 + lo];
      for (int r = 0; r < 8; ++r) {
        float vv = fmaxf(acc[r] + bias, 0.f);
        actA[w][(r + hi * 8) * 64 + nt * 16 + lo] = f2bf(vv);
      }
    }
  }
  // ---- heads: actA -> logits (16x16, cols 0-4 ly, 5-9 lh, 10-14 ld) ----
  {
    v16bf a0 = loadA(actA[w], 64, 0);
    v16bf a1 = loadA(actA[w], 64, 32);
    v8f acc = {};
    acc = __builtin_amdgcn_wmma_f32_16x16x32_bf16(false, a0, false,
            loadB(WHs, 0), (short)0, acc, false, false);
    acc = __builtin_amdgcn_wmma_f32_16x16x32_bf16(false, a1, false,
            loadB(WHs, 1), (short)0, acc, false, false);
    float bias = (lo < 5) ? by[lo] : (lo < 10) ? bh[lo - 5]
               : (lo < 15) ? bd[lo - 10] : 0.f;
    for (int r = 0; r < 8; ++r)
      lgt[w][(r + hi * 8) * 16 + lo] = acc[r] + bias;
  }
  // ---- gumbel-hard forward == argmax(logits + g) ----
  if (lam < 16) {
    int tok = s * 16 + lam;
    const float* row = &lgt[w][lam * 16];
    const float* gHt = gH + (size_t)tok * 5;
    const float* gdt = gd + (size_t)tok * 5;
    float best = row[5] + gHt[0]; int bi = 0;
    for (int i = 1; i < 5; ++i) {
      float vv = row[5 + i] + gHt[i];
      if (vv > best) { best = vv; bi = i; }
    }
    ih[tok] = bi;
    best = row[10] + gdt[0]; bi = 0;
    for (int i = 1; i < 5; ++i) {
      float vv = row[10 + i] + gdt[i];
      if (vv > best) { best = vv; bi = i; }
    }
    id[tok] = bi;
  }
  if (lam < 2) {
    int bl = s * 2 + lam;
    float sum[5] = {0.f, 0.f, 0.f, 0.f, 0.f};
    for (int r = 0; r < 8; ++r)
      for (int i = 0; i < 5; ++i)
        sum[i] += lgt[w][(lam * 8 + r) * 16 + i];
    const float* gyt = gy + (size_t)bl * 5;
    float best = sum[0] * 0.125f + gyt[0]; int bi = 0;
    for (int i = 1; i < 5; ++i) {
      float vv = sum[i] * 0.125f + gyt[i];
      if (vv > best) { best = vv; bi = i; }
    }
    iy[bl] = bi;
  }
}

// ============================================================================
// K4a: quantize H -> H_q (one complex pair per thread, idx per (b,l,k))
// ============================================================================
__global__ void k4_quantH(const float* __restrict__ H, const int* __restrict__ ih,
                          const float* __restrict__ sH, float* __restrict__ out) {
  size_t p = (size_t)blockIdx.x * 256 + threadIdx.x;   // 8388608 pairs
  int k = (int)(p & 7);
  size_t bl = p >> 9;
  int idx = ih[bl * 8 + k];
  float2 x = ((const float2*)H)[p];
  float2 o = { quant1(x.x, idx, sH), quant1(x.y, idx, sH) };
  ((float2*)(out + OFF_HQ))[p] = o;
}

// K4b: quantize y -> y_q (idx per (b,l))
__global__ void k4_quantY(const float* __restrict__ y, const int* __restrict__ iy,
                          const float* __restrict__ sY, float* __restrict__ out) {
  size_t p = (size_t)blockIdx.x * 256 + threadIdx.x;   // 1048576 pairs
  int idx = iy[p >> 6];
  float2 x = ((const float2*)y)[p];
  float2 o = { quant1(x.x, idx, sY), quant1(x.y, idx, sY) };
  ((float2*)(out + OFF_YQ))[p] = o;
}

// K4c: quantize v + emit bit-link outputs (per token)
__global__ void k4_quantV(const float* __restrict__ v, const int* __restrict__ iy,
                          const int* __restrict__ ih, const int* __restrict__ id,
                          const float* __restrict__ sD, float* __restrict__ out) {
  int tok = blockIdx.x * 256 + threadIdx.x;            // 131072
  int bl = tok >> 3;
  int iyv = iy[bl], ihv = ih[tok], idv = id[tok];
  float2 x = ((const float2*)v)[tok];
  float2 o = { quant1(x.x, idv, sD), quant1(x.y, idv, sD) };
  ((float2*)(out + OFF_VQ))[tok] = o;
  // BIT_VALS[i] = 4*i ; b_y = 2*64*4*iy/8 ; b_H = 2*64*4*ih ; b_d = 2*4*id
  float byv = 64.f * (float)iyv;
  float bHv = 512.f * (float)ihv;
  float bdv = 8.f * (float)idv;
  out[OFF_BY + tok]  = byv;
  out[OFF_BH + tok]  = bHv;
  out[OFF_BD + tok]  = bdv;
  out[OFF_TOT + tok] = byv + bHv + bdv;
}

// ============================================================================
extern "C" void kernel_launch(void* const* d_in, const int* in_sizes, int n_in,
                              void* d_out, int out_size, void* d_ws, size_t ws_size,
                              hipStream_t stream) {
  const float* v    = (const float*)d_in[0];
  const float* H    = (const float*)d_in[1];
  const float* y    = (const float*)d_in[2];
  const float* snr  = (const float*)d_in[3];
  const float* gy   = (const float*)d_in[4];
  const float* gH   = (const float*)d_in[5];
  const float* gd   = (const float*)d_in[6];
  const float* W1   = (const float*)d_in[7];
  const float* b1   = (const float*)d_in[8];
  const float* W2   = (const float*)d_in[9];
  const float* b2   = (const float*)d_in[10];
  const float* W3   = (const float*)d_in[11];
  const float* b3   = (const float*)d_in[12];
  const float* Wy   = (const float*)d_in[13];
  const float* by   = (const float*)d_in[14];
  const float* Wh   = (const float*)d_in[15];
  const float* bh   = (const float*)d_in[16];
  const float* Wd   = (const float*)d_in[17];
  const float* bd   = (const float*)d_in[18];
  const float* sY   = (const float*)d_in[19];
  const float* sH   = (const float*)d_in[20];
  const float* sD   = (const float*)d_in[21];
  float* out = (float*)d_out;

  char* ws = (char*)d_ws;
  float*    hp  = (float*)(ws + WS_HP);
  float*    avg = (float*)(ws + WS_AVG);
  float*    ypw = (float*)(ws + WS_YPW);
  ushort_t* X   = (ushort_t*)(ws + WS_X);
  ushort_t* W1s = (ushort_t*)(ws + WS_W1S);
  ushort_t* W2s = (ushort_t*)(ws + WS_W2S);
  ushort_t* W3s = (ushort_t*)(ws + WS_W3S);
  ushort_t* WHs = (ushort_t*)(ws + WS_WHS);
  int*      iy  = (int*)(ws + WS_IY);
  int*      ih  = (int*)(ws + WS_IH);
  int*      id  = (int*)(ws + WS_ID);

  k0_prep<<<1, 256, 0, stream>>>(W1, W2, W3, Wy, Wh, Wd, avg, W1s, W2s, W3s, WHs);
  k1_power<<<NBL, 128, 0, stream>>>(H, y, hp, avg, ypw);
  k2_features<<<NTOK / 256, 256, 0, stream>>>(v, snr, hp, avg, ypw, X);
  k3_mlp<<<NTOK / 16 / 8, 256, 0, stream>>>(X, W1s, W2s, W3s, WHs,
                                            b1, b2, b3, by, bh, bd,
                                            gy, gH, gd, iy, ih, id);
  k4_quantH<<<8388608 / 256, 256, 0, stream>>>(H, ih, sH, out);
  k4_quantY<<<1048576 / 256, 256, 0, stream>>>(y, iy, sY, out);
  k4_quantV<<<NTOK / 256, 256, 0, stream>>>(v, iy, ih, id, sD, out);
}
